// POESurvivalPerTimeBin_89395449299514
// MI455X (gfx1250) — matrix operand order repaired and checked by reading, not compiled, not run on hardware
//
#include <hip/hip_runtime.h>

#define DD 256            // in_dim
#define MM 8              // n_models
#define TT 32             // T
#define T1 33             // bins
#define NTOT 264          // T1*MM (gate GEMM N)
#define EPSV 1e-12f

#define ROWS 64           // rows per workgroup
#define NTHREADS 256      // 8 waves (wave32)

#define WG_STRIDE 264     // bf16 per Wg row  (256 + 8 pad, 16B-aligned rows)
#define SX_STRIDE 264     // bf16 per x row
#define SRAW_STRIDE 276   // f32 per raw row (pad to dodge bank conflicts)
#define SLOG_STRIDE 260   // f32 per logits row (TDM pad: 256 + 4 dwords)

typedef __attribute__((ext_vector_type(16))) __bf16 v16bf;
typedef __attribute__((ext_vector_type(8)))  __bf16 v8bf;
typedef __attribute__((ext_vector_type(8)))  float  v8f;
typedef __attribute__((ext_vector_type(4)))  unsigned int u32x4;
typedef __attribute__((ext_vector_type(8)))  int i32x8;
typedef __attribute__((ext_vector_type(4)))  int i32x4;

union BF16x16 {
  v16bf v;
  struct { v8bf lo, hi; } p;
};

__device__ __forceinline__ unsigned short f2bf(float f) {
  unsigned u = __builtin_bit_cast(unsigned, f);
  u += 0x7FFFu + ((u >> 16) & 1u);          // round-to-nearest-even
  return (unsigned short)(u >> 16);
}
__device__ __forceinline__ unsigned pk2bf(float a, float b) {
  return (unsigned)f2bf(a) | ((unsigned)f2bf(b) << 16);
}
__device__ __forceinline__ unsigned lds_off(const void* p) {
  return (unsigned)(unsigned long long)(uintptr_t)p;   // flat->LDS: addr[31:0]
}

__global__ __launch_bounds__(NTHREADS)
void poe_wmma_kernel(const float* __restrict__ x,
                     const float* __restrict__ logits,
                     const float* __restrict__ Wg,
                     const float* __restrict__ bg,
                     float* __restrict__ out, int Bn) {
  __shared__ __align__(16) unsigned short sWg[272 * WG_STRIDE];   // 143,616 B
  __shared__ __align__(16) unsigned short sX [ROWS * SX_STRIDE];  //  33,792 B
  __shared__ __align__(16) float sRaw[ROWS * SRAW_STRIDE];        //  70,656 B
  __shared__ __align__(16) float sLog[ROWS * SLOG_STRIDE];        //  66,560 B
  __shared__ __align__(16) float sBg[NTOT];                       //   1,056 B

  const int tid  = threadIdx.x;
  const int wave = tid >> 5;
  const int row0 = blockIdx.x * ROWS;

  // ---- wave 0: TDM copy of the logits tile (64 rows x 256 f32) into sLog,
  //      with LDS padding 256+4 dwords per row (stride 260) done by the TDM ----
  if (wave == 0) {
    const unsigned long long ga =
        (unsigned long long)(uintptr_t)(logits + (size_t)row0 * (MM * TT));
    u32x4 g0;
    g0[0] = 1u;                                   // count=1, user load
    g0[1] = lds_off(&sLog[0]);                    // lds_addr
    g0[2] = (unsigned)(ga & 0xffffffffu);         // global_addr[31:0]
    g0[3] = (unsigned)((ga >> 32) & 0x01ffffffu)  // global_addr[56:32]
            | (2u << 30);                         // type=2 ("image")
    i32x8 g1;
    g1[0] = (int)((2u << 16)      // data_size = 4B
                  | (1u << 20)    // pad_enable
                  | (7u << 22)    // pad_interval: 256 dwords
                  | (3u << 25));  // pad_amount:   4 dwords
    g1[1] = (int)(256u << 16);    // tensor_dim0 = 256 (low16 at [63:48])
    g1[2] = (int)(64u << 16);     // tensor_dim0 hi=0, tensor_dim1 = 64
    g1[3] = (int)(256u << 16);    // tensor_dim1 hi=0, tile_dim0 = 256
    g1[4] = 64;                   // tile_dim1 = 64, tile_dim2 = 0
    g1[5] = 256;                  // tensor_dim0_stride = 256 (low32)
    g1[6] = 0;
    g1[7] = 0;
    i32x4 gz4 = {0, 0, 0, 0};
    i32x8 gz8 = {0, 0, 0, 0, 0, 0, 0, 0};
    __builtin_amdgcn_tensor_load_to_lds(g0, g1, gz4, gz4, gz8, 0);
  } else if (wave == 1) {
    // ---- wave 1: async-copy bg (264 f32 = 66 x b128) into sBg ----
    const int l = tid & 31;
    for (int i = l; i < NTOT / 4; i += 32) {
      unsigned dst = lds_off(&sBg[i * 4]);
      unsigned long long ga = (unsigned long long)(uintptr_t)(bg + i * 4);
      asm volatile("global_load_async_to_lds_b128 %0, %1, off"
                   :: "v"(dst), "v"(ga) : "memory");
    }
  }

  // Speculatively warm the next block's x tile in L2 while we convert.
  if (row0 + ROWS < Bn) {
    const char* nx = (const char*)(x + (size_t)(row0 + ROWS) * DD);
    for (int i = tid; i < 128; i += NTHREADS)
      __builtin_prefetch(nx + i * 512, 0, 1);
  }

  // ---- stage Wg: f32 -> bf16, row-major [n][k] (this IS the B^T layout) ----
  const float4* wg4 = (const float4*)Wg;
  for (int i = tid; i < NTOT * (DD / 4); i += NTHREADS) {
    int nr = i >> 6, c4 = i & 63;
    float4 v = wg4[i];
    uint2 pv; pv.x = pk2bf(v.x, v.y); pv.y = pk2bf(v.z, v.w);
    *(uint2*)(void*)&sWg[nr * WG_STRIDE + c4 * 4] = pv;
  }
  for (int i = tid; i < 8 * (WG_STRIDE / 2); i += NTHREADS)  // zero pad rows 264..271
    ((unsigned*)sWg)[NTOT * (WG_STRIDE / 2) + i] = 0u;

  // ---- stage x tile: f32 -> bf16 ----
  const float4* x4 = (const float4*)x + (size_t)row0 * (DD / 4);
  for (int i = tid; i < ROWS * (DD / 4); i += NTHREADS) {
    int r = i >> 6, c4 = i & 63;
    float4 v = x4[i];
    uint2 pv; pv.x = pk2bf(v.x, v.y); pv.y = pk2bf(v.z, v.w);
    *(uint2*)(void*)&sX[r * SX_STRIDE + c4 * 4] = pv;
  }

  // Complete our own async transfers before the block-wide barrier.
  if (wave == 0) {
    __builtin_amdgcn_s_wait_tensorcnt(0);
  } else if (wave == 1) {
    asm volatile("s_wait_asynccnt 0x0" ::: "memory");
  }
  __syncthreads();

  // ---- GEMM: raw[r][n] = sum_k xbf[r][k] * wgbf[n][k]  (bf16 WMMA) ----
  {
    const int lane = tid & 31;
    const int lrow = lane & 15;
    const int hi   = lane >> 4;
    const int slab = wave & 3;                 // 16-row slab
    const int tstart = (wave >> 2) ? 9 : 0;    // split 17 N-tiles across 2 waves
    const int tend   = (wave >> 2) ? 17 : 9;

    // A fragments: 16x32 bf16 per K-step; lane (lrow,hi) reads K chunks
    // [kb..kb+7] and [kb+16..kb+23] of row slab*16+lrow (per ISA layout).
    BF16x16 a[8];
    const int arow = slab * 16 + lrow;
#pragma unroll
    for (int kk = 0; kk < 8; ++kk) {
      const int kb = kk * 32 + hi * 8;
      a[kk].p.lo = *(const v8bf*)(const void*)&sX[arow * SX_STRIDE + kb];
      a[kk].p.hi = *(const v8bf*)(const void*)&sX[arow * SX_STRIDE + kb + 16];
    }

    for (int tile = tstart; tile < tend; ++tile) {
      v8f c = {0.f, 0.f, 0.f, 0.f, 0.f, 0.f, 0.f, 0.f};
      const int nr = tile * 16 + lrow;   // B column = N index (Wg row)
#pragma unroll
      for (int kk = 0; kk < 8; ++kk) {
        BF16x16 bfr;                      // K chunks [kb..kb+7],[kb+16..kb+23] of column nr
        const int kb = kk * 32 + hi * 8;
        bfr.p.lo = *(const v8bf*)(const void*)&sWg[nr * WG_STRIDE + kb];
        bfr.p.hi = *(const v8bf*)(const void*)&sWg[nr * WG_STRIDE + kb + 16];
        c = __builtin_amdgcn_wmma_f32_16x16x32_bf16(
                false, a[kk].v, false, bfr.v, (short)0, c, false, false);
      }
      const int col = tile * 16 + lrow;  // C layout: VGPR i -> row i+8*hi
#pragma unroll
      for (int i = 0; i < 8; ++i)
        sRaw[(slab * 16 + hi * 8 + i) * SRAW_STRIDE + col] = c[i];
    }
  }

  __syncthreads();

  // ---- epilogue: one thread per row ----
  if (tid < ROWS) {
    const int r = tid;
    const size_t b = (size_t)row0 + r;

    float S_prev[MM], wm[MM];
#pragma unroll
    for (int m = 0; m < MM; ++m) { S_prev[m] = 1.f; wm[m] = 0.f; }

    float flp[T1];
#pragma unroll
    for (int t1 = 0; t1 < T1; ++t1) {
      float e[MM], mx = -3.0e38f;
#pragma unroll
      for (int m = 0; m < MM; ++m) {
        float rv = sRaw[r * SRAW_STRIDE + t1 * MM + m] + sBg[t1 * MM + m];
        e[m] = rv; mx = fmaxf(mx, rv);
      }
      float den = 0.f;
#pragma unroll
      for (int m = 0; m < MM; ++m) { e[m] = __expf(e[m] - mx); den += e[m]; }
      const float inv = 1.f / den;
      float acc = 0.f;
#pragma unroll
      for (int m = 0; m < MM; ++m) {
        const float w = e[m] * inv;
        wm[m] += w * (1.0f / (float)T1);
        float lp;
        if (t1 < TT) {                         // hazard bins
          const float lg = sLog[r * SLOG_STRIDE + m * TT + t1];
          const float h  = 1.f / (1.f + __expf(-lg));
          const float pe = fmaxf(S_prev[m] * h, EPSV);
          lp = __logf(pe + EPSV);
          S_prev[m] *= fmaxf(1.f - h, EPSV);
        } else {                               // tail bin
          lp = __logf(fmaxf(S_prev[m], EPSV) + EPSV);
        }
        acc += w * lp;
      }
      flp[t1] = acc;
    }

    // softmax over 33 fused log-probs
    float fm = -3.0e38f;
#pragma unroll
    for (int t1 = 0; t1 < T1; ++t1) fm = fmaxf(fm, flp[t1]);
    float fden = 0.f;
#pragma unroll
    for (int t1 = 0; t1 < T1; ++t1) { flp[t1] = __expf(flp[t1] - fm); fden += flp[t1]; }
    const float finv = 1.f / fden;

    // full dist -> hazards
    float cdf = 0.f;
    float* outH = out + b * (size_t)TT;
#pragma unroll
    for (int t = 0; t < TT; ++t) {
      const float pe = fmaxf(flp[t] * finv, EPSV);
      const float Sp = (t == 0) ? 1.f : fmaxf(1.f - cdf, EPSV);
      float hz = pe / Sp;
      hz = fminf(fmaxf(hz, EPSV), 1.f - EPSV);
      outH[t] = hz;
      cdf += pe;
    }
    float* outW = out + (size_t)Bn * TT + b * (size_t)MM;
#pragma unroll
    for (int m = 0; m < MM; ++m) outW[m] = wm[m];
  }
}

extern "C" void kernel_launch(void* const* d_in, const int* in_sizes, int n_in,
                              void* d_out, int out_size, void* d_ws, size_t ws_size,
                              hipStream_t stream) {
  const float* x      = (const float*)d_in[0];
  const float* logits = (const float*)d_in[1];
  const float* Wg     = (const float*)d_in[2];
  const float* bg     = (const float*)d_in[3];
  float* out = (float*)d_out;

  const int Bn = in_sizes[0] / DD;          // 65536
  dim3 grid(Bn / ROWS), block(NTHREADS);
  hipLaunchKernelGGL(poe_wmma_kernel, grid, block, 0, stream,
                     x, logits, Wg, bg, out, Bn);
}
